// Interpolate_82128364634645
// MI455X (gfx1250) — compile-verified, matrix operand order checked by reading, not run on hardware
//
#include <hip/hip_runtime.h>
#include <hip/hip_bf16.h>
#include <stdint.h>

typedef __attribute__((ext_vector_type(16))) _Float16 v16h;
typedef __attribute__((ext_vector_type(8)))  float    v8f;
typedef __attribute__((ext_vector_type(8)))  int      v8i;

#define NPTS 4096
#define CCH  256
#define NB   4

static __device__ __forceinline__ unsigned short f2h_bits(float x) {
  _Float16 h = (_Float16)x;
  return __builtin_bit_cast(unsigned short, h);
}
static __device__ __forceinline__ float h2f_lo(unsigned u) {
  return (float)__builtin_bit_cast(_Float16, (unsigned short)(u & 0xFFFF));
}
static __device__ __forceinline__ float h2f_hi(unsigned u) {
  return (float)__builtin_bit_cast(_Float16, (unsigned short)(u >> 16));
}
static __device__ __forceinline__ unsigned umin(unsigned a, unsigned b) { return a < b ? a : b; }
static __device__ __forceinline__ unsigned umax(unsigned a, unsigned b) { return a > b ? a : b; }
// branchless insert of key into sorted top-3 (k0 <= k1 <= k2): 5 VALU, no EXEC writes
static __device__ __forceinline__ void ins3(unsigned &k0, unsigned &k1, unsigned &k2,
                                            unsigned key) {
  unsigned m0 = umax(k0, key);
  k0 = umin(k0, key);
  unsigned m1 = umax(k1, m0);
  k1 = umin(k1, m0);
  k2 = umin(k2, m1);
}

// ---------------- K1: per-(batch,cloud) coordinate max -> step ----------------
__global__ __launch_bounds__(256)
void k_coord_max(const int* __restrict__ src_c, const int* __restrict__ tgt_c,
                 int* __restrict__ steps) {
  const int b = blockIdx.x >> 1, cloud = blockIdx.x & 1;
  const int* c = (cloud ? tgt_c : src_c) + (size_t)b * NPTS * 3;
  int mx = 0;
  for (int i = threadIdx.x; i < NPTS * 3; i += 256) {
    int v = c[i];
    mx = (v > mx) ? v : mx;
  }
  __shared__ int red[256];
  red[threadIdx.x] = mx;
  __syncthreads();
  for (int s = 128; s > 0; s >>= 1) {
    if ((int)threadIdx.x < s) {
      int v = red[threadIdx.x + s];
      if (v > red[threadIdx.x]) red[threadIdx.x] = v;
    }
    __syncthreads();
  }
  if (threadIdx.x == 0) steps[blockIdx.x] = red[0] + 1;
}

// ---------------- K2: stable sort (bitonic on (key<<12)|idx) + pack f16 coords ----
__global__ __launch_bounds__(1024)
void k_sort_pack(const int* __restrict__ src_c, const int* __restrict__ tgt_c,
                 const int* __restrict__ steps,
                 int* __restrict__ si,
                 unsigned* __restrict__ sxy, unsigned* __restrict__ sz0,
                 unsigned* __restrict__ txy, unsigned* __restrict__ tz0,
                 float* __restrict__ tt2) {
  const int b = blockIdx.x >> 1, cloud = blockIdx.x & 1;
  const int* c = (cloud ? tgt_c : src_c) + (size_t)b * NPTS * 3;
  const long long step = steps[blockIdx.x];
  __shared__ unsigned long long keys[NPTS];
  for (int i = threadIdx.x; i < NPTS; i += 1024) {
    long long x = c[3 * i], y = c[3 * i + 1], z = c[3 * i + 2];
    unsigned long long key = (unsigned long long)(x + y * step + z * step * step);
    keys[i] = (key << 12) | (unsigned long long)i;  // unique keys => stable
  }
  __syncthreads();
  for (unsigned k = 2; k <= NPTS; k <<= 1) {
    for (unsigned j = k >> 1; j > 0; j >>= 1) {
      for (unsigned p = threadIdx.x; p < NPTS / 2; p += 1024) {
        unsigned i = ((p & ~(j - 1)) << 1) | (p & (j - 1));
        unsigned q = i | j;
        bool asc = ((i & k) == 0);
        unsigned long long a = keys[i], bb = keys[q];
        if ((a > bb) == asc) { keys[i] = bb; keys[q] = a; }
      }
      __syncthreads();
    }
  }
  for (int i = threadIdx.x; i < NPTS; i += 1024) {
    int p = (int)(keys[i] & 0xFFFULL);
    float x = (float)c[3 * p], y = (float)c[3 * p + 1], z = (float)c[3 * p + 2];
    unsigned xy = (unsigned)f2h_bits(x) | ((unsigned)f2h_bits(y) << 16);
    unsigned z0 = (unsigned)f2h_bits(z);
    size_t o = (size_t)b * NPTS + i;
    if (cloud == 0) {
      si[o] = p; sxy[o] = xy; sz0[o] = z0;
    } else {
      txy[o] = xy; tz0[o] = z0; tt2[o] = x * x + y * y + z * z;
    }
  }
}

// ---------------- K3: 3-NN via f16 WMMA distance tiles ----------------
// Block = 8 waves, each wave owns 16 target rows; loops over 256 source tiles,
// unrolled x2 so the 2nd WMMA issue + tile-0 insert ALU hides the WMMA->VALU
// hazard window of the 1st.
__global__ __launch_bounds__(256)
void k_nn_wmma(const unsigned* __restrict__ sxy, const unsigned* __restrict__ sz0,
               const unsigned* __restrict__ txy, const unsigned* __restrict__ tz0,
               const float* __restrict__ tt2,
               int* __restrict__ nn_idx, float* __restrict__ nn_w) {
  const int b = blockIdx.y;
  const int wave = threadIdx.x >> 5;
  const int lane = threadIdx.x & 31;
  const int mbase = blockIdx.x * 128 + wave * 16;
  const size_t bO = (size_t)b * NPTS;

  __shared__ unsigned s_xy[NPTS];            // 16 KB
  __shared__ unsigned s_z0[NPTS];            // 16 KB
  __shared__ unsigned s_mrg[8][32][24];      // 24 KB

  for (int i = threadIdx.x; i < NPTS; i += 256) {
    s_xy[i] = sxy[bO + i];
    s_z0[i] = sz0[bO + i];
  }
  __syncthreads();

  const bool lo16 = (lane < 16);

  // A (16x32 f16): lane<16 holds row M=lane; dword0 = (x,y) halves, dword1 = (z,0).
  // Lanes 16..31 hold K=8..15,24..31 -> all zero.
  v8i ai = {};
  {
    unsigned xy = lo16 ? txy[bO + mbase + (lane & 15)] : 0u;
    unsigned z0 = lo16 ? tz0[bO + mbase + (lane & 15)] : 0u;
    ai[0] = (int)xy;
    ai[1] = (int)z0;
  }
  const v16h A = __builtin_bit_cast(v16h, ai);

  // |t|^2 for the 8 D-rows this lane sees (rows 0..7 for lanes<16, 8..15 else)
  float t2r[8];
  {
    const int half8 = (lane >> 4) << 3;
#pragma unroll
    for (int r = 0; r < 8; ++r) t2r[r] = tt2[bO + mbase + r + half8];
  }

  unsigned k0[8], k1[8], k2[8];
#pragma unroll
  for (int r = 0; r < 8; ++r) { k0[r] = 0xFFFFFFFFu; k1[r] = 0xFFFFFFFFu; k2[r] = 0xFFFFFFFFu; }

  const int ncol = lane & 15;
  v8i bi0 = {};   // persistent: dwords 2..7 stay zero, no per-tile re-zeroing
  v8i bi1 = {};

  for (int nb = 0; nb < NPTS / 16; nb += 2) {
    const int n0 = nb * 16;

    unsigned xyA = s_xy[n0 + ncol];
    unsigned z0A = s_z0[n0 + ncol];
    unsigned xyB = s_xy[n0 + 16 + ncol];
    unsigned z0B = s_z0[n0 + 16 + ncol];

    float s2A, s2B;
    {
      float sx = h2f_lo(xyA), sy = h2f_hi(xyA), sz = h2f_lo(z0A);
      s2A = sx * sx + sy * sy + sz * sz;
    }
    {
      float sx = h2f_lo(xyB), sy = h2f_hi(xyB), sz = h2f_lo(z0B);
      s2B = sx * sx + sy * sy + sz * sz;
    }

    // B (32x16 f16): lanes 0..15 hold K=0..7,16..23 for column N=lane; lanes 16..31 zero.
    bi0[0] = (int)(lo16 ? xyA : 0u);
    bi0[1] = (int)(lo16 ? z0A : 0u);
    bi1[0] = (int)(lo16 ? xyB : 0u);
    bi1[1] = (int)(lo16 ? z0B : 0u);

    v8f cz = {};
    v8f d0 = __builtin_amdgcn_wmma_f32_16x16x32_f16(false, A, false,
                                                    __builtin_bit_cast(v16h, bi0),
                                                    (short)0, cz, false, false);
    v8f d1 = __builtin_amdgcn_wmma_f32_16x16x32_f16(false, A, false,
                                                    __builtin_bit_cast(v16h, bi1),
                                                    (short)0, cz, false, false);

    const unsigned nidx0 = (unsigned)(n0 + ncol);
    const unsigned nidx1 = nidx0 + 16u;
#pragma unroll
    for (int r = 0; r < 8; ++r) {
      float d2 = t2r[r] + s2A - 2.0f * d0[r];  // exact small-int arithmetic
      unsigned key = (((unsigned)d2) << 12) | nidx0;
      ins3(k0[r], k1[r], k2[r], key);
    }
#pragma unroll
    for (int r = 0; r < 8; ++r) {
      float d2 = t2r[r] + s2B - 2.0f * d1[r];
      unsigned key = (((unsigned)d2) << 12) | nidx1;
      ins3(k0[r], k1[r], k2[r], key);
    }
  }

#pragma unroll
  for (int r = 0; r < 8; ++r) {
    s_mrg[wave][lane][r * 3 + 0] = k0[r];
    s_mrg[wave][lane][r * 3 + 1] = k1[r];
    s_mrg[wave][lane][r * 3 + 2] = k2[r];
  }
  __syncthreads();

  if (lane < 16) {
    unsigned b0 = 0xFFFFFFFFu, b1 = 0xFFFFFFFFu, b2 = 0xFFFFFFFFu;
    const int rr = lane & 7;
    const int l0 = (lane < 8) ? 0 : 16;
    for (int l = l0; l < l0 + 16; ++l) {
      ins3(b0, b1, b2, s_mrg[wave][l][rr * 3 + 0]);
      ins3(b0, b1, b2, s_mrg[wave][l][rr * 3 + 1]);
      ins3(b0, b1, b2, s_mrg[wave][l][rr * 3 + 2]);
    }
    const int m = mbase + lane;
    float d0 = (float)(b0 >> 12), d1 = (float)(b1 >> 12), d2 = (float)(b2 >> 12);
    float r0 = 1.0f / (d0 + 1e-8f);
    float r1 = 1.0f / (d1 + 1e-8f);
    float r2 = 1.0f / (d2 + 1e-8f);
    float sum = r0 + r1 + r2;
    size_t o = ((size_t)b * NPTS + m) * 3;
    nn_idx[o + 0] = (int)(b0 & 0xFFFu);
    nn_idx[o + 1] = (int)(b1 & 0xFFFu);
    nn_idx[o + 2] = (int)(b2 & 0xFFFu);
    nn_w[o + 0] = r0 / sum;
    nn_w[o + 1] = r1 / sum;
    nn_w[o + 2] = r2 / sum;
  }
}

// ---------------- K4: IDW interp + LayerNorm + exact GELU ----------------
__global__ __launch_bounds__(256)
void k_interp_ln_gelu(const float* __restrict__ feats, const int* __restrict__ si,
                      const int* __restrict__ nn_idx, const float* __restrict__ nn_w,
                      const float* __restrict__ gamma, const float* __restrict__ beta,
                      float* __restrict__ out) {
  const int b = blockIdx.y;
  const int m = blockIdx.x;
  const int c = threadIdx.x;
  const size_t bm = (size_t)b * NPTS + m;
  const size_t o3 = bm * 3;
  float acc = 0.0f;
#pragma unroll
  for (int t = 0; t < 3; ++t) {
    int idx = nn_idx[o3 + t];
    float w = nn_w[o3 + t];
    int srow = si[(size_t)b * NPTS + idx];              // un-permute: f_sorted[idx]
    acc += w * feats[((size_t)b * NPTS + srow) * CCH + c];
  }
  float s1 = acc, sq = acc * acc;
#pragma unroll
  for (int off = 16; off > 0; off >>= 1) {
    s1 += __shfl_xor(s1, off, 32);
    sq += __shfl_xor(sq, off, 32);
  }
  __shared__ float rs1[8], rsq[8];
  const int w = threadIdx.x >> 5, ln = threadIdx.x & 31;
  if (ln == 0) { rs1[w] = s1; rsq[w] = sq; }
  __syncthreads();
  float S1 = 0.0f, SQ = 0.0f;
#pragma unroll
  for (int i = 0; i < 8; ++i) { S1 += rs1[i]; SQ += rsq[i]; }
  float mu = S1 * (1.0f / CCH);
  float var = SQ * (1.0f / CCH) - mu * mu;
  float y = (acc - mu) * rsqrtf(var + 1e-6f) * gamma[c] + beta[c];
  out[bm * CCH + c] = 0.5f * y * (1.0f + erff(y * 0.70710678118654752440f));
}

// ---------------- launch ----------------
extern "C" void kernel_launch(void* const* d_in, const int* in_sizes, int n_in,
                              void* d_out, int out_size, void* d_ws, size_t ws_size,
                              hipStream_t stream) {
  (void)in_sizes; (void)n_in; (void)out_size; (void)ws_size;
  const float* feats = (const float*)d_in[0];
  const int*   src_c = (const int*)d_in[1];
  const int*   tgt_c = (const int*)d_in[2];
  const float* gamma = (const float*)d_in[3];
  const float* beta  = (const float*)d_in[4];
  float* out = (float*)d_out;

  char* w = (char*)d_ws;
  auto alloc = [&](size_t bytes) -> char* {
    char* p = w;
    w += (bytes + 255) & ~(size_t)255;
    return p;
  };
  int*      d_steps = (int*)alloc(8 * sizeof(int));
  int*      d_si  = (int*)alloc((size_t)NB * NPTS * sizeof(int));
  unsigned* d_sxy = (unsigned*)alloc((size_t)NB * NPTS * 4);
  unsigned* d_sz0 = (unsigned*)alloc((size_t)NB * NPTS * 4);
  unsigned* d_txy = (unsigned*)alloc((size_t)NB * NPTS * 4);
  unsigned* d_tz0 = (unsigned*)alloc((size_t)NB * NPTS * 4);
  float*    d_tt2 = (float*)alloc((size_t)NB * NPTS * 4);
  int*      d_nni = (int*)alloc((size_t)NB * NPTS * 3 * 4);
  float*    d_nnw = (float*)alloc((size_t)NB * NPTS * 3 * 4);

  k_coord_max<<<8, 256, 0, stream>>>(src_c, tgt_c, d_steps);
  k_sort_pack<<<8, 1024, 0, stream>>>(src_c, tgt_c, d_steps, d_si,
                                      d_sxy, d_sz0, d_txy, d_tz0, d_tt2);
  k_nn_wmma<<<dim3(NPTS / 128, NB), 256, 0, stream>>>(d_sxy, d_sz0, d_txy, d_tz0,
                                                      d_tt2, d_nni, d_nnw);
  k_interp_ln_gelu<<<dim3(NPTS, NB), 256, 0, stream>>>(feats, d_si, d_nni, d_nnw,
                                                       gamma, beta, out);
}
